// VectorQuantizer_64029372449365
// MI455X (gfx1250) — matrix operand order repaired and checked by reading, not compiled, not run on hardware
//
#include <hip/hip_runtime.h>

// ---------------- problem constants ----------------
#define NROWS   131072      // 32 * 4096
#define DIM     64
#define NCODES  1024
#define NTILES  (NCODES / 16)   // 64 code tiles of 16
#define NCHUNK  2               // DIM 64 = 2 chunks of K=32 (bf16 WMMA)
#define Q_ELEMS (NROWS * DIM)   // 8388608
#define LOSS_SCALE (1.25f / (float)(NROWS * DIM))

typedef __attribute__((ext_vector_type(16))) __bf16 v16bf;
typedef __attribute__((ext_vector_type(8)))  float  v8f;

union BFrag { unsigned u[8]; v16bf v; };

// round-to-nearest-even f32 -> bf16 (as u16)
__device__ __forceinline__ unsigned bf16rne(float f) {
    unsigned u = __float_as_uint(f);
    return (u + 0x7FFFu + ((u >> 16) & 1u)) >> 16;
}
__device__ __forceinline__ unsigned pkbf(float lo, float hi) {
    return bf16rne(lo) | (bf16rne(hi) << 16);
}

// ------------------------------------------------------------------
// Kernel 1: pack codebook into bf16 WMMA B-fragment layout.
// B (32x16 bf16, per chunk): lane l element e  <->  K = e + 16*(l>>4),
// N = l&15.  dword i packs elements {2i, 2i+1}.
// ws layout: bfrag[((t*2 + c)*32 + lane)*8 + i]
// ------------------------------------------------------------------
__global__ __launch_bounds__(256) void vq_pack(const float* __restrict__ cb,
                                               unsigned* __restrict__ bfrag) {
    int gid = blockIdx.x * 256 + threadIdx.x;          // 0 .. 4095
    if (gid >= NTILES * NCHUNK * 32) return;
    int t    = gid >> 6;
    int c    = (gid >> 5) & 1;
    int lane = gid & 31;
    int code = t * 16 + (lane & 15);
    int dimbase = c * 32 + 16 * (lane >> 4);           // 16 consecutive dims
    const float* src = cb + code * DIM + dimbase;
    unsigned* dst = bfrag + gid * 8;
#pragma unroll
    for (int i = 0; i < 8; ++i)
        dst[i] = pkbf(src[2 * i], src[2 * i + 1]);
}

// ------------------------------------------------------------------
// Kernel 2: codebook squared norms + zero the loss accumulator.
// ------------------------------------------------------------------
__global__ __launch_bounds__(256) void vq_enorm(const float* __restrict__ cb,
                                                float* __restrict__ enorm,
                                                float* __restrict__ out_loss) {
    int gid = blockIdx.x * 256 + threadIdx.x;          // 0 .. 1023
    if (gid == 0) out_loss[0] = 0.0f;
    if (gid >= NCODES) return;
    const float* r = cb + gid * DIM;
    float s = 0.0f;
#pragma unroll 8
    for (int j = 0; j < DIM; ++j) s = fmaf(r[j], r[j], s);
    enorm[gid] = s;
}

// ------------------------------------------------------------------
// Kernel 3: main VQ kernel.
// 512 threads = 16 waves; each wave owns 2 row-groups of 16 -> 512 rows/block.
// Argmin keys: distance f32 with code index packed into low 10 mantissa bits,
// reduced with v_min_f32 (3 VALU/element epilogue, no VCC chains).
// ------------------------------------------------------------------
__global__ __launch_bounds__(512) void vq_main(const float* __restrict__ x,
                                               const float* __restrict__ cb,
                                               const unsigned* __restrict__ bfrag,
                                               const float* __restrict__ enorm,
                                               float* __restrict__ out_q,
                                               float* __restrict__ out_codes,
                                               float* __restrict__ out_loss) {
    __shared__ unsigned sB[NTILES * NCHUNK * 32 * 8];  // 128 KB packed B frags
    __shared__ float    sEn[NCODES];                   // 4 KB norms

    const int tid = threadIdx.x;

    // cooperative stage of codebook fragments + norms into LDS
    for (int i = tid; i < NTILES * NCHUNK * 32 * 8; i += 512) sB[i] = bfrag[i];
    for (int i = tid; i < NCODES; i += 512)                   sEn[i] = enorm[i];
    __syncthreads();

    const int wave    = tid >> 5;
    const int lane    = tid & 31;
    const int half    = lane >> 4;        // 0 or 1
    const int ln      = lane & 15;        // N column within tile / row-in-half
    const int rowbase = blockIdx.x * 512 + wave * 32;   // 2 groups of 16 rows

    // ---- build A fragments (contiguous K: element s <-> k = 16*half + s, same as B) ----
    const int koff = half * 16;
    BFrag a[2][NCHUNK];
#pragma unroll
    for (int g = 0; g < 2; ++g) {
        const float* xr = x + (rowbase + g * 16 + ln) * DIM;
#pragma unroll
        for (int c = 0; c < NCHUNK; ++c) {
            const int b1 = c * 32 + koff;
            float4 f0 = *(const float4*)(xr + b1);
            float4 f1 = *(const float4*)(xr + b1 + 4);
            float4 g0 = *(const float4*)(xr + b1 + 8);
            float4 g1 = *(const float4*)(xr + b1 + 12);
            a[g][c].u[0] = pkbf(f0.x, f0.y);
            a[g][c].u[1] = pkbf(f0.z, f0.w);
            a[g][c].u[2] = pkbf(f1.x, f1.y);
            a[g][c].u[3] = pkbf(f1.z, f1.w);
            a[g][c].u[4] = pkbf(g0.x, g0.y);
            a[g][c].u[5] = pkbf(g0.z, g0.w);
            a[g][c].u[6] = pkbf(g1.x, g1.y);
            a[g][c].u[7] = pkbf(g1.z, g1.w);
        }
    }

    float bestk[2][8];
#pragma unroll
    for (int g = 0; g < 2; ++g)
#pragma unroll
        for (int v = 0; v < 8; ++v) bestk[g][v] = __uint_as_float(0x7F800000u);

    // B-fragment LDS fetch: dword base = lane*8 + tile*512 (+256 for chunk 1)
    auto ldB = [&](int t, BFrag& c0, BFrag& c1) {
        const unsigned* p = &sB[lane * 8 + t * 512];
        *(int4*)&c0.u[0] = *(const int4*)(p);
        *(int4*)&c0.u[4] = *(const int4*)(p + 4);
        *(int4*)&c1.u[0] = *(const int4*)(p + 256);
        *(int4*)&c1.u[4] = *(const int4*)(p + 260);
    };

    // one tile: 4 WMMAs (2 row-groups x 2 K-chunks), prefetch next B, min-pack
    auto step = [&](int t, BFrag& b0, BFrag& b1, BFrag& n0, BFrag& n1,
                    float cen, float& nen) {
        v8f acc0 = {}, acc1 = {};
        acc0 = __builtin_amdgcn_wmma_f32_16x16x32_bf16(
            false, a[0][0].v, false, b0.v, (short)0, acc0, false, false);
        acc1 = __builtin_amdgcn_wmma_f32_16x16x32_bf16(
            false, a[1][0].v, false, b0.v, (short)0, acc1, false, false);
        acc0 = __builtin_amdgcn_wmma_f32_16x16x32_bf16(
            false, a[0][1].v, false, b1.v, (short)0, acc0, false, false);
        acc1 = __builtin_amdgcn_wmma_f32_16x16x32_bf16(
            false, a[1][1].v, false, b1.v, (short)0, acc1, false, false);

        const int tn = (t + 1) & (NTILES - 1);         // prefetch (hides LDS lat.)
        ldB(tn, n0, n1);
        nen = sEn[tn * 16 + ln];

        const unsigned code = (unsigned)(t * 16 + ln);
#pragma unroll
        for (int v = 0; v < 8; ++v) {
            float d0 = fmaf(-2.0f, acc0[v], cen);      // ||e||^2 - 2 x.e
            float d1 = fmaf(-2.0f, acc1[v], cen);
            float k0 = __uint_as_float((__float_as_uint(d0) & 0xFFFFFC00u) | code);
            float k1 = __uint_as_float((__float_as_uint(d1) & 0xFFFFFC00u) | code);
            bestk[0][v] = fminf(bestk[0][v], k0);
            bestk[1][v] = fminf(bestk[1][v], k1);
        }
    };

    // ping-pong double buffer, 2x unrolled so buffer selection is static
    BFrag bA0, bA1, bB0, bB1;
    float enA, enB;
    ldB(0, bA0, bA1);
    enA = sEn[ln];
    for (int t = 0; t < NTILES; t += 2) {
        step(t,     bA0, bA1, bB0, bB1, enA, enB);
        step(t + 1, bB0, bB1, bA0, bA1, enB, enA);
    }

    // ---- reduce argmin keys across the 16 lanes of each N-group ----
#pragma unroll
    for (int m = 1; m < 16; m <<= 1) {
#pragma unroll
        for (int g = 0; g < 2; ++g)
#pragma unroll
            for (int v = 0; v < 8; ++v)
                bestk[g][v] = fminf(bestk[g][v], __shfl_xor(bestk[g][v], m, 32));
    }
    // C layout: lanes 0-15 hold rows base+v, lanes 16-31 rows base+8+v

    float err = 0.0f;
#pragma unroll
    for (int g = 0; g < 2; ++g) {
        const int gbase = rowbase + g * 16 + 8 * half;
        if (ln == 0) {
#pragma unroll
            for (int v = 0; v < 8; ++v)
                out_codes[gbase + v] =
                    (float)(__float_as_uint(bestk[g][v]) & 1023u);
        }
        // gather quantized rows (f32 codebook, L2-resident) + squared error
#pragma unroll
        for (int v = 0; v < 8; ++v) {
            const int row  = gbase + v;
            const int code = (int)(__float_as_uint(bestk[g][v]) & 1023u);
            float4 q  = *(const float4*)(cb + code * DIM + ln * 4);
            float4 xi = *(const float4*)(x  + row  * DIM + ln * 4);
            *(float4*)(out_q + row * DIM + ln * 4) = q;  // straight-through fwd
            float dx = xi.x - q.x, dy = xi.y - q.y;
            float dz = xi.z - q.z, dw = xi.w - q.w;
            err = fmaf(dx, dx, err); err = fmaf(dy, dy, err);
            err = fmaf(dz, dz, err); err = fmaf(dw, dw, err);
        }
    }
#pragma unroll
    for (int m = 1; m < 32; m <<= 1) err += __shfl_xor(err, m, 32);
    if (lane == 0) atomicAdd(out_loss, err * LOSS_SCALE);
}

// ------------------------------------------------------------------
extern "C" void kernel_launch(void* const* d_in, const int* in_sizes, int n_in,
                              void* d_out, int out_size, void* d_ws, size_t ws_size,
                              hipStream_t stream) {
    const float* x  = (const float*)d_in[0];   // [32,4096,64] f32
    const float* cb = (const float*)d_in[1];   // [1024,64]    f32

    unsigned* bfrag = (unsigned*)d_ws;                              // 128 KB
    float*    enorm = (float*)((char*)d_ws + NTILES * NCHUNK * 32 * 8 * 4); // 4 KB

    float* out_q     = (float*)d_out;                 // 8388608 elems
    float* out_codes = out_q + Q_ELEMS;               // 131072 elems
    float* out_loss  = out_codes + NROWS;             // 1 elem

    vq_pack <<<16, 256, 0, stream>>>(cb, bfrag);
    vq_enorm<<<4,  256, 0, stream>>>(cb, enorm, out_loss);
    vq_main <<<NROWS / 512, 512, 0, stream>>>(x, cb, bfrag, enorm,
                                              out_q, out_codes, out_loss);
}
